// Phi3SeerAttention_29661044146340
// MI455X (gfx1250) — compile-verified, matrix operand order checked by reading, not run on hardware
//
#include <hip/hip_runtime.h>
#include <hip/hip_bf16.h>

// ---------------------------------------------------------------------------
// Phi3 SeerAttention forward for MI455X (gfx1250, wave32, WMMA).
// All GEMM-like math runs through v_wmma_f32_16x16x32_bf16 with f32 accum.
// GEMM tile fills use CDNA5 async global->LDS copies (ASYNCcnt) so the
// next-tile prefetch has no VGPR staging and fully overlaps the WMMAs.
// ---------------------------------------------------------------------------

typedef __bf16 bf16_t;
typedef __attribute__((ext_vector_type(16))) __bf16 v16bf;
typedef __attribute__((ext_vector_type(8)))  float  v8f;

#define S_LEN 2048
#define HID   3072
#define NH    32
#define NKV   8
#define HD    96
#define OPW   4608     // H*D + 2*HK*D
#define BLKSZ 64
#define NBLK  32
#define GHID  128

__device__ __forceinline__ bf16_t f2bf(float f) {
  unsigned int u = __builtin_bit_cast(unsigned int, f);
  unsigned int r = (u + 0x7FFFu + ((u >> 16) & 1u)) >> 16;  // round-nearest-even
  unsigned short s = (unsigned short)r;
  return __builtin_bit_cast(bf16_t, s);
}

// Async 16-byte global -> LDS copy (no VGPR staging, tracked by ASYNCcnt).
// lds_addr: AS(3) byte offset (low 32 bits of a generic pointer to __shared__).
__device__ __forceinline__ void async_cp16(unsigned lds_addr, const bf16_t* gptr) {
  asm volatile("global_load_async_to_lds_b128 %0, %1, off"
               :: "v"(lds_addr), "v"(gptr)
               : "memory");
}
__device__ __forceinline__ void async_wait0() {
  asm volatile("s_wait_asynccnt 0x0" ::: "memory");
}

// Load one 16x32 bf16 WMMA operand fragment.
// Source is row-major with row stride `rs` (elements); lane L<16 supplies
// row L, K = {hi*8..hi*8+7, 16+hi*8..16+hi*8+7} where hi = L>=16 — matching
// the CDNA5 16-bit A-matrix VGPR layout. Two 16-byte loads per lane.
__device__ __forceinline__ v16bf frag_ld(const bf16_t* p, int rs) {
  const int lane = threadIdx.x & 31;
  const bf16_t* q = p + (lane & 15) * rs + ((lane >> 4) << 3);
  v16bf r;
  *((float4*)&r)       = *((const float4*)q);
  *(((float4*)&r) + 1) = *((const float4*)(q + 16));
  return r;
}

// Row-wise reductions over the 16 lanes that hold one C-matrix row.
__device__ __forceinline__ float rowmax16(float x) {
#pragma unroll
  for (int d = 1; d < 16; d <<= 1) x = fmaxf(x, __shfl_xor(x, d, 32));
  return x;
}
__device__ __forceinline__ float rowsum16(float x) {
#pragma unroll
  for (int d = 1; d < 16; d <<= 1) x += __shfl_xor(x, d, 32);
  return x;
}

// ---------------------------------------------------------------------------
// Elementwise f32 -> bf16 cast, 8 elements/thread, b128 in / b128 out.
// ---------------------------------------------------------------------------
__global__ __launch_bounds__(256) void cast_bf16_kernel(
    const float* __restrict__ X, bf16_t* __restrict__ Y, size_t n) {
  const size_t i = ((size_t)blockIdx.x * 256 + threadIdx.x) * 8;
  if (i >= n) return;
  float4 a = *(const float4*)(X + i);
  float4 b = *(const float4*)(X + i + 4);
  alignas(16) bf16_t v[8];
  v[0] = f2bf(a.x); v[1] = f2bf(a.y); v[2] = f2bf(a.z); v[3] = f2bf(a.w);
  v[4] = f2bf(b.x); v[5] = f2bf(b.y); v[6] = f2bf(b.z); v[7] = f2bf(b.w);
  *(float4*)(Y + i) = *(const float4*)v;
}

// ---------------------------------------------------------------------------
// LDS-tiled weight transpose + cast: W[K][N] (f32) -> WT[N][K] (bf16).
// 32x32 tile, padded LDS to avoid bank conflicts; both sides coalesced.
// K, N multiples of 32.
// ---------------------------------------------------------------------------
__global__ __launch_bounds__(256) void wT_bf16_tiled(
    const float* __restrict__ W, bf16_t* __restrict__ WT, int K, int N) {
  __shared__ float tile[32][33];
  const int kb = blockIdx.y * 32, nb = blockIdx.x * 32;
  const int tx = threadIdx.x & 31, ty = threadIdx.x >> 5;  // ty = 0..7
#pragma unroll
  for (int r = ty; r < 32; r += 8)
    tile[r][tx] = W[(size_t)(kb + r) * N + nb + tx];
  __syncthreads();
#pragma unroll
  for (int r = ty; r < 32; r += 8)
    WT[(size_t)(nb + r) * K + kb + tx] = f2bf(tile[tx][r]);
}

// ---------------------------------------------------------------------------
// WMMA GEMM: C[M][N] (f32) = A[M][K] x BT[N][K] (both bf16), K%32==0,
// M%128==0, N%128==0. Block = 256 threads = 8 waves (4x2), tile 128x128x32,
// double-buffered LDS filled by async global->LDS b128 copies, wave tile
// 32x64 (2x4 WMMA accumulators).
// ---------------------------------------------------------------------------
__global__ __launch_bounds__(256) void gemm_wmma_bf16(
    const bf16_t* __restrict__ A, const bf16_t* __restrict__ BT,
    float* __restrict__ C, int M, int N, int K) {
  __shared__ alignas(16) bf16_t As[2][128][32];
  __shared__ alignas(16) bf16_t Bs[2][128][32];
  const int tid  = threadIdx.x;
  const int m0   = blockIdx.y * 128;
  const int n0   = blockIdx.x * 128;
  const int wave = tid >> 5;
  const int lane = tid & 31;
  const int wm   = (wave >> 1) * 32;   // wave row offset in tile
  const int wn   = (wave & 1) * 64;    // wave col offset in tile
  const int fr   = tid >> 1;           // fill row 0..127
  const int fc   = (tid & 1) * 16;     // fill col 0 or 16
  v8f acc[2][4] = {};
  const int nk = K >> 5;

  const bf16_t* arow = &A[(size_t)(m0 + fr) * K + fc];
  const bf16_t* brow = &BT[(size_t)(n0 + fr) * K + fc];

  // Fill tile 0 (async, then drain + barrier).
  {
    const unsigned la = (unsigned)(size_t)&As[0][fr][fc];
    const unsigned lb = (unsigned)(size_t)&Bs[0][fr][fc];
    async_cp16(la,      arow);
    async_cp16(la + 16, arow + 8);
    async_cp16(lb,      brow);
    async_cp16(lb + 16, brow + 8);
  }
  async_wait0();
  __syncthreads();

  for (int kt = 0; kt < nk; ++kt) {
    const int cur = kt & 1;
    if (kt + 1 < nk) {  // async prefetch of next tile, overlaps the WMMAs below
      const int k1 = (kt + 1) << 5, nxt = cur ^ 1;
      const unsigned la = (unsigned)(size_t)&As[nxt][fr][fc];
      const unsigned lb = (unsigned)(size_t)&Bs[nxt][fr][fc];
      async_cp16(la,      arow + k1);
      async_cp16(la + 16, arow + k1 + 8);
      async_cp16(lb,      brow + k1);
      async_cp16(lb + 16, brow + k1 + 8);
    }
    if (kt + 2 < nk) {  // L2 warming for the tile after next
      __builtin_prefetch(arow + ((kt + 2) << 5), 0, 1);
      __builtin_prefetch(brow + ((kt + 2) << 5), 0, 1);
    }
    v16bf af[2], bfq[4];
#pragma unroll
    for (int mt = 0; mt < 2; ++mt) af[mt] = frag_ld(&As[cur][wm + mt * 16][0], 32);
#pragma unroll
    for (int nt = 0; nt < 4; ++nt) bfq[nt] = frag_ld(&Bs[cur][wn + nt * 16][0], 32);
#pragma unroll
    for (int mt = 0; mt < 2; ++mt)
#pragma unroll
      for (int nt = 0; nt < 4; ++nt)
        acc[mt][nt] = __builtin_amdgcn_wmma_f32_16x16x32_bf16(
            false, af[mt], false, bfq[nt], (short)0, acc[mt][nt], false, false);
    async_wait0();     // own async writes into nxt complete
    __syncthreads();   // publish to consumer waves / cover WAR on cur
  }

  // C layout: VGPR j holds row (j + 8*(lane>=16)), cols striped over lanes 0..15
  const int cn = lane & 15;
  const int mb = (lane >> 4) << 3;
#pragma unroll
  for (int mt = 0; mt < 2; ++mt)
#pragma unroll
    for (int nt = 0; nt < 4; ++nt)
#pragma unroll
      for (int j = 0; j < 8; ++j) {
        int r = m0 + wm + mt * 16 + mb + j;
        int c = n0 + wn + nt * 16 + cn;
        C[(size_t)r * N + c] = acc[mt][nt][j];
      }
}

// ---------------------------------------------------------------------------
// RoPE + split into head-major bf16 tensors:
//   qbf[h][s][96], kbf[hk][s][96] (roped), vT[hk][96][s] (transposed)
// ---------------------------------------------------------------------------
__global__ __launch_bounds__(256) void rope_split_kernel(
    const float* __restrict__ qkv, const float* __restrict__ cosb,
    const float* __restrict__ sinb, bf16_t* __restrict__ qbf,
    bf16_t* __restrict__ kbf, bf16_t* __restrict__ vT) {
  const int s = blockIdx.y;
  const int c = blockIdx.x * 256 + threadIdx.x;  // 0..4607
  const float x = qkv[(size_t)s * OPW + c];
  if (c < NH * HD) {
    const int h = c / HD, d = c - h * HD;
    const int pd = (d < 48) ? (c + 48) : (c - 48);
    const float sgn = (d < 48) ? -1.f : 1.f;
    const float rx = qkv[(size_t)s * OPW + pd];
    const float val = x * cosb[s * HD + d] + sgn * rx * sinb[s * HD + d];
    qbf[((size_t)h * S_LEN + s) * HD + d] = f2bf(val);
  } else if (c < NH * HD + NKV * HD) {
    const int kk = c - NH * HD;
    const int hk = kk / HD, d = kk - hk * HD;
    const int pd = (d < 48) ? (c + 48) : (c - 48);
    const float sgn = (d < 48) ? -1.f : 1.f;
    const float rx = qkv[(size_t)s * OPW + pd];
    const float val = x * cosb[s * HD + d] + sgn * rx * sinb[s * HD + d];
    kbf[((size_t)hk * S_LEN + s) * HD + d] = f2bf(val);
  } else {
    const int vv = c - NH * HD - NKV * HD;
    const int hk = vv / HD, d = vv - hk * HD;
    vT[((size_t)hk * HD + d) * S_LEN + s] = f2bf(x);
  }
}

// ---------------------------------------------------------------------------
// k-gate: pool (mean,max over BLK) of pre-RoPE k, project by gate_wk[192][128]
// ---------------------------------------------------------------------------
__global__ __launch_bounds__(128) void kgate_kernel(
    const float* __restrict__ qkv, const float* __restrict__ gate_wk,
    float* __restrict__ k_gate) {
  const int n = blockIdx.x;   // block index
  const int h = blockIdx.y;   // kv head
  __shared__ float smean[HD], smax[HD];
  const int tid = threadIdx.x;
  if (tid < HD) {
    float su = 0.f, mx = -1e30f;
    for (int t = 0; t < BLKSZ; ++t) {
      float v = qkv[(size_t)(n * BLKSZ + t) * OPW + NH * HD + h * HD + tid];
      su += v;
      mx = fmaxf(mx, v);
    }
    smean[tid] = su * (1.f / BLKSZ);
    smax[tid]  = mx;
  }
  __syncthreads();
  float acc = 0.f;
  for (int d = 0; d < HD; ++d) {
    acc += smean[d] * gate_wk[d * GHID + tid];
    acc += smax[d]  * gate_wk[(HD + d) * GHID + tid];
  }
  k_gate[((size_t)n * NKV + h) * GHID + tid] = acc;
}

// ---------------------------------------------------------------------------
// q-gate: pool pre-RoPE q over (4 heads in group x BLK tokens), project.
// ---------------------------------------------------------------------------
__global__ __launch_bounds__(128) void qgate_kernel(
    const float* __restrict__ qkv, const float* __restrict__ gate_wq,
    float* __restrict__ q_gate) {
  const int n = blockIdx.x;
  const int h = blockIdx.y;
  __shared__ float pool[HD];
  const int tid = threadIdx.x;
  if (tid < HD) {
    float su = 0.f;
    for (int t = 0; t < BLKSZ; ++t)
      for (int g4 = 0; g4 < 4; ++g4)
        su += qkv[(size_t)(n * BLKSZ + t) * OPW + (h * 4 + g4) * HD + tid];
    pool[tid] = su * (1.f / (BLKSZ * 4));
  }
  __syncthreads();
  float acc = 0.f;
  for (int d = 0; d < HD; ++d) acc += pool[d] * gate_wq[d * GHID + tid];
  q_gate[((size_t)n * NKV + h) * GHID + tid] = acc;
}

// ---------------------------------------------------------------------------
// Gate logits + causal block softmax + threshold -> 32-bit mask per (hk,qb).
// One thread per (kv head, q block).
// ---------------------------------------------------------------------------
__global__ __launch_bounds__(256) void mask_kernel(
    const float* __restrict__ q_gate, const float* __restrict__ k_gate,
    unsigned int* __restrict__ mask32) {
  const int tid = threadIdx.x;
  const int h  = tid >> 5;   // kv head 0..7
  const int qb = tid & 31;   // q block 0..31
  float logits[NBLK];
  const float gscale = 0.08838834764831845f;  // 128^-0.5
  for (int kb = 0; kb <= qb; ++kb) {
    float dot = 0.f;
    for (int g = 0; g < GHID; ++g)
      dot += q_gate[((size_t)qb * NKV + h) * GHID + g] *
             k_gate[((size_t)kb * NKV + h) * GHID + g];
    logits[kb] = dot * gscale;
  }
  float m = -1e30f;
  for (int kb = 0; kb <= qb; ++kb) m = fmaxf(m, logits[kb]);
  float ssum = 0.f;
  for (int kb = 0; kb <= qb; ++kb) {
    logits[kb] = __expf(logits[kb] - m);
    ssum += logits[kb];
  }
  const float inv = 1.f / ssum;
  unsigned int bits = 1u << qb;  // diagonal always visible
  for (int kb = 0; kb <= qb; ++kb)
    if (logits[kb] * inv >= 0.03f) bits |= (1u << kb);
  mask32[h * NBLK + qb] = bits;
}

// ---------------------------------------------------------------------------
// Block-sparse flash attention. Block = (q block, head), 4 waves x 16 q rows.
// Scores via 3 WMMAs per 16-col tile (K = 96); online softmax via 16-lane
// shuffles on the C layout; P staged through LDS into A-fragment layout;
// P @ V via 12 WMMAs per visited kv block (vT gives contiguous B fragments).
// ---------------------------------------------------------------------------
__global__ __launch_bounds__(128) void attn_kernel(
    const bf16_t* __restrict__ qbf, const bf16_t* __restrict__ kbf,
    const bf16_t* __restrict__ vT, const unsigned int* __restrict__ mask32,
    bf16_t* __restrict__ attn_out) {
  const int qb   = blockIdx.x;
  const int h    = blockIdx.y;
  const int hk   = h >> 2;
  const int w    = threadIdx.x >> 5;
  const int lane = threadIdx.x & 31;
  const int q0   = qb * BLKSZ + w * 16;
  const float scale = 0.1020620726159658f;  // 96^-0.5
  __shared__ alignas(16) bf16_t lds_p[4][16][BLKSZ];

  // Q fragments: loop-invariant, loaded once per wave.
  v16bf aq[3];
  const bf16_t* qbase = qbf + ((size_t)h * S_LEN + q0) * HD;
#pragma unroll
  for (int kc = 0; kc < 3; ++kc) aq[kc] = frag_ld(qbase + kc * 32, HD);

  float m[8], l[8];
  v8f o[6] = {};
#pragma unroll
  for (int j = 0; j < 8; ++j) { m[j] = -1e30f; l[j] = 0.f; }

  const unsigned int mask = mask32[hk * NBLK + qb];  // uniform -> EXEC stays full
  const int cn = lane & 15;
  const int mb = (lane >> 4) << 3;

  for (int kb = 0; kb <= qb; ++kb) {
    if (!((mask >> kb) & 1u)) continue;
    const int kv0 = kb * BLKSZ;

    // S = Q x K^T for this 16x64 strip (4 n-tiles, 3 k-steps each)
    v8f s[4];
#pragma unroll
    for (int nt = 0; nt < 4; ++nt) {
      v8f acc = {};
      const bf16_t* kbase = kbf + ((size_t)hk * S_LEN + kv0 + nt * 16) * HD;
#pragma unroll
      for (int kc = 0; kc < 3; ++kc) {
        v16bf bk = frag_ld(kbase + kc * 32, HD);
        acc = __builtin_amdgcn_wmma_f32_16x16x32_bf16(false, aq[kc], false, bk,
                                                      (short)0, acc, false, false);
      }
      s[nt] = acc;
    }

    // Online softmax (row = mb + j across the 16 lanes of a half-wave)
    const bool diag = (kb == qb);
    float fac[8];
#pragma unroll
    for (int j = 0; j < 8; ++j) {
      const int qrow = q0 + mb + j;
      float rmax = -1e30f;
#pragma unroll
      for (int nt = 0; nt < 4; ++nt) {
        float v = s[nt][j] * scale;
        if (diag && (kv0 + nt * 16 + cn) > qrow) v = -1e30f;  // token causal
        s[nt][j] = v;
        rmax = fmaxf(rmax, v);
      }
      rmax = rowmax16(rmax);
      const float mn = fmaxf(m[j], rmax);
      fac[j] = __expf(m[j] - mn);
      m[j] = mn;
      float rsum = 0.f;
#pragma unroll
      for (int nt = 0; nt < 4; ++nt) {
        const float p = __expf(s[nt][j] - mn);
        s[nt][j] = p;
        rsum += p;
      }
      rsum = rowsum16(rsum);
      l[j] = l[j] * fac[j] + rsum;
    }
#pragma unroll
    for (int t = 0; t < 6; ++t)
#pragma unroll
      for (int j = 0; j < 8; ++j) o[t][j] *= fac[j];

    // Stage P (bf16) into LDS in row-major 16x64 so it can be re-read as
    // A fragments (C layout -> A layout crosses lanes; LDS does the shuffle).
#pragma unroll
    for (int nt = 0; nt < 4; ++nt)
#pragma unroll
      for (int j = 0; j < 8; ++j)
        lds_p[w][mb + j][nt * 16 + cn] = f2bf(s[nt][j]);
    asm volatile("s_wait_dscnt 0" ::: "memory");  // stores visible to own wave's loads

    // O += P x V  (6 d-tiles, 2 k-steps over the 64 kv positions)
#pragma unroll
    for (int t = 0; t < 6; ++t) {
      v8f acc = o[t];
      const bf16_t* vbase = vT + ((size_t)hk * HD + t * 16) * S_LEN + kv0;
#pragma unroll
      for (int kc = 0; kc < 2; ++kc) {
        v16bf ap = frag_ld(&lds_p[w][0][0] + kc * 32, BLKSZ);
        v16bf bv = frag_ld(vbase + kc * 32, S_LEN);
        acc = __builtin_amdgcn_wmma_f32_16x16x32_bf16(false, ap, false, bv,
                                                      (short)0, acc, false, false);
      }
      o[t] = acc;
    }
    asm volatile("s_wait_dscnt 0" ::: "memory");  // drain before next overwrite
  }

  // Epilogue: normalize and write bf16 attention output [S][H*D]
#pragma unroll
  for (int j = 0; j < 8; ++j) {
    const float inv = 1.f / l[j];
    const int row = q0 + mb + j;
#pragma unroll
    for (int t = 0; t < 6; ++t)
      attn_out[(size_t)row * HID + h * HD + t * 16 + cn] = f2bf(o[t][j] * inv);
  }
}

// ---------------------------------------------------------------------------
// Host-side pipeline
// ---------------------------------------------------------------------------
extern "C" void kernel_launch(void* const* d_in, const int* in_sizes, int n_in,
                              void* d_out, int out_size, void* d_ws, size_t ws_size,
                              hipStream_t stream) {
  const float* hidden  = (const float*)d_in[0];
  const float* cosb    = (const float*)d_in[1];
  const float* sinb    = (const float*)d_in[2];
  const float* qkv_w   = (const float*)d_in[3];
  const float* o_w     = (const float*)d_in[4];
  const float* gate_wq = (const float*)d_in[5];
  const float* gate_wk = (const float*)d_in[6];
  float* out = (float*)d_out;
  (void)in_sizes; (void)n_in; (void)out_size; (void)ws_size;

  char* ws = (char*)d_ws;
  auto carve = [&](size_t bytes) {
    void* p = (void*)ws;
    ws += (bytes + 255) & ~(size_t)255;
    return p;
  };
  float*  qkv       = (float*)carve((size_t)S_LEN * OPW * 4);        // fused qkv (f32)
  bf16_t* hidden_bf = (bf16_t*)carve((size_t)S_LEN * HID * 2);       // cast once
  bf16_t* qkv_wT    = (bf16_t*)carve((size_t)OPW * HID * 2);         // [N][K] bf16
  bf16_t* o_wT      = (bf16_t*)carve((size_t)HID * HID * 2);         // [N][K] bf16
  bf16_t* qbf       = (bf16_t*)carve((size_t)NH  * S_LEN * HD * 2);  // roped q
  bf16_t* kbf       = (bf16_t*)carve((size_t)NKV * S_LEN * HD * 2);  // roped k
  bf16_t* vTb       = (bf16_t*)carve((size_t)NKV * HD * S_LEN * 2);  // v transposed
  bf16_t* attn_bf   = (bf16_t*)carve((size_t)S_LEN * HID * 2);       // attn out
  float*  k_gate    = (float*)carve((size_t)NBLK * NKV * GHID * 4);
  float*  q_gate    = (float*)carve((size_t)NBLK * NKV * GHID * 4);
  unsigned int* mask32 = (unsigned int*)carve((size_t)NKV * NBLK * 4);

  // One-time bf16 conversions (vectorized / LDS-tiled transpose).
  {
    size_t n = (size_t)S_LEN * HID;  // 6.29M elems, 8/thread
    cast_bf16_kernel<<<(unsigned)(n / 8 / 256), 256, 0, stream>>>(hidden, hidden_bf, n);
  }
  wT_bf16_tiled<<<dim3(OPW / 32, HID / 32), 256, 0, stream>>>(qkv_w, qkv_wT, HID, OPW);
  wT_bf16_tiled<<<dim3(HID / 32, HID / 32), 256, 0, stream>>>(o_w, o_wT, HID, HID);

  // qkv = hidden @ qkv_w   (WMMA bf16, f32 out)
  gemm_wmma_bf16<<<dim3(OPW / 128, S_LEN / 128), 256, 0, stream>>>(
      hidden_bf, qkv_wT, qkv, S_LEN, OPW, HID);

  rope_split_kernel<<<dim3(OPW / 256, S_LEN), 256, 0, stream>>>(
      qkv, cosb, sinb, qbf, kbf, vTb);

  kgate_kernel<<<dim3(NBLK, NKV), 128, 0, stream>>>(qkv, gate_wk, k_gate);
  qgate_kernel<<<dim3(NBLK, NKV), 128, 0, stream>>>(qkv, gate_wq, q_gate);
  mask_kernel<<<1, 256, 0, stream>>>(q_gate, k_gate, mask32);

  attn_kernel<<<dim3(NBLK, NH), 128, 0, stream>>>(qbf, kbf, vTb, mask32, attn_bf);

  // out = attn @ o_w   (WMMA bf16, f32 out -> d_out)
  gemm_wmma_bf16<<<dim3(HID / 128, S_LEN / 128), 256, 0, stream>>>(
      attn_bf, o_wT, out, S_LEN, HID, HID);
}
